// WanSelfAttention_76965813944615
// MI455X (gfx1250) — matrix operand order repaired and checked by reading, not compiled, not run on hardware
//
#include <hip/hip_runtime.h>
#include <hip/hip_bf16.h>
#include <cstdint>

// ---------------------------------------------------------------------------
// WanSelfAttention for MI455X (gfx1250, wave32, WMMA + Tensor Data Mover)
//   DIM=1536, HEADS=12, HD=128, L=3584 (F=8,H=16,W=28)
// ---------------------------------------------------------------------------

#define DIM   1536
#define HEADS 12
#define HD    128
#define LQ    3584
#define H_    16
#define W_    28
#define EPS   1e-6f

typedef __attribute__((ext_vector_type(16))) _Float16     v16h;
typedef __attribute__((ext_vector_type(8)))  float        v8f;
typedef __attribute__((ext_vector_type(4)))  unsigned int u32x4;
typedef __attribute__((ext_vector_type(8)))  unsigned int u32x8;

// ---------------------------------------------------------------------------
// Tensor Data Mover: 2D tile (f16 elements) global -> LDS, ISA §8.3/§8.4.
//   tile_d0 = elements per row, tile_d1 = rows, stride = row stride (elems)
//   pad_iv/pad_am: LDS padding codes (interval 2^(iv+1) DW, amount am+1 DW)
// ---------------------------------------------------------------------------
__device__ inline void tdm_load_2d_f16(unsigned lds_off, const void* gptr,
                                       unsigned tile_d0, unsigned tile_d1,
                                       unsigned stride_elts,
                                       unsigned pad_iv, unsigned pad_am) {
  const unsigned long long ga = (unsigned long long)(uintptr_t)gptr;
  u32x4 g0;
  g0[0] = 1u;                                   // count=1 (valid), no gather
  g0[1] = lds_off;                              // lds_addr (bytes)
  g0[2] = (unsigned)(ga & 0xffffffffu);         // global_addr[31:0]
  g0[3] = (unsigned)((ga >> 32) & 0x01ffffffu)  // global_addr[56:32]
        | (2u << 30);                           // type=2 (image)
  u32x8 g1;
  g1[0] = (1u << 16)                            // data_size=1 -> 2 bytes
        | (1u << 20)                            // pad_enable
        | (pad_iv << 22) | (pad_am << 25);
  g1[1] = (tile_d0 & 0xffffu) << 16;            // tensor_dim0[15:0] (== tile_d0)
  g1[2] = (tile_d0 >> 16) | ((tile_d1 & 0xffffu) << 16);  // dim0 hi | tensor_dim1 lo
  g1[3] = (tile_d1 >> 16) | (tile_d0 << 16);    // dim1 hi | tile_dim0
  g1[4] = tile_d1 & 0xffffu;                    // tile_dim1 | tile_dim2=0
  g1[5] = stride_elts;                          // tensor_dim0_stride[31:0]
  g1[6] = 0u;
  g1[7] = 0u;
  asm volatile("tensor_load_to_lds %0, %1" :: "s"(g0), "s"(g1) : "memory");
}

__device__ inline unsigned lds_off_of(const void* p) {
  return (unsigned)(uintptr_t)p;                // LDS aperture: offset in [31:0]
}

// ---------------------------------------------------------------------------
// Fragment builders per CDNA5 ISA §7.12.2 (wave32 layouts)
// ---------------------------------------------------------------------------
__device__ inline v16h load_a_frag(const _Float16* __restrict__ smem, int stride, int lane) {
  const int m  = lane & 15;
  const int kb = (lane < 16) ? 0 : 8;
  const _Float16* row = smem + m * stride;
  v16h a;
#pragma unroll
  for (int v = 0; v < 4; ++v) {
    a[2 * v]     = row[kb + 2 * v];
    a[2 * v + 1] = row[kb + 2 * v + 1];
  }
#pragma unroll
  for (int v = 0; v < 4; ++v) {
    a[8 + 2 * v] = row[kb + 16 + 2 * v];
    a[9 + 2 * v] = row[kb + 16 + 2 * v + 1];
  }
  return a;
}

// B 32x16 from LDS stored [n][k]: B[k][n] = smem[n*stride + k]  (coalesced b128s)
__device__ inline v16h load_b_frag_nk(const _Float16* __restrict__ smem, int stride, int lane) {
  const int n  = lane & 15;
  const int kb = (lane < 16) ? 0 : 8;
  const _Float16* row = smem + n * stride;
  v16h b;
#pragma unroll
  for (int v = 0; v < 4; ++v) {
    b[2 * v]     = row[kb + 2 * v];
    b[2 * v + 1] = row[kb + 2 * v + 1];
  }
#pragma unroll
  for (int v = 0; v < 4; ++v) {
    b[8 + 2 * v] = row[kb + 16 + 2 * v];
    b[9 + 2 * v] = row[kb + 16 + 2 * v + 1];
  }
  return b;
}

// ---------------------------------------------------------------------------
// Stage 0: f32 -> f16 convert
// ---------------------------------------------------------------------------
__global__ __launch_bounds__(256) void cvt_f32_f16(const float* __restrict__ in,
                                                   _Float16* __restrict__ out, int n) {
  for (int i = blockIdx.x * 256 + threadIdx.x; i < n; i += gridDim.x * 256)
    out[i] = (_Float16)in[i];
}

// ---------------------------------------------------------------------------
// GEMM: C[M,N] = A[M,K](f16) @ W[N,K](f16)^T + bias(f32)
// 128x128 block tile, 4 waves, 64x64 per wave, TDM double-buffered staging.
// ---------------------------------------------------------------------------
#define GK   32
#define GSTR 40   // 32 data + 8 pad halves = 20 DW rows (TDM pad: 16DW + 4DW)

__global__ __launch_bounds__(128) void gemm_f16(
    const _Float16* __restrict__ A, const _Float16* __restrict__ W,
    const float* __restrict__ bias, float* __restrict__ C,
    int M, int N, int K) {
  __shared__ _Float16 As[2][128][GSTR];
  __shared__ _Float16 Bs[2][128][GSTR];

  const int tid  = threadIdx.x;
  const int lane = tid & 31;
  const int wave = tid >> 5;
  const int m0 = blockIdx.x * 128;
  const int n0 = blockIdx.y * 128;
  const int mw = (wave >> 1) * 64;
  const int nw = (wave & 1) * 64;

  v8f acc[16] = {};

  if (wave == 0) {
    tdm_load_2d_f16(lds_off_of(&As[0][0][0]), &A[(size_t)m0 * K], GK, 128, K, 3, 3);
    tdm_load_2d_f16(lds_off_of(&Bs[0][0][0]), &W[(size_t)n0 * K], GK, 128, K, 3, 3);
  }

  int buf = 0;
  for (int kk = 0; kk < K; kk += GK, buf ^= 1) {
    __syncthreads();  // buf^1 free to overwrite
    const bool more = (kk + GK) < K;
    if (wave == 0 && more) {
      tdm_load_2d_f16(lds_off_of(&As[buf ^ 1][0][0]), &A[(size_t)m0 * K + kk + GK],
                      GK, 128, K, 3, 3);
      tdm_load_2d_f16(lds_off_of(&Bs[buf ^ 1][0][0]), &W[(size_t)n0 * K + kk + GK],
                      GK, 128, K, 3, 3);
    }
    if (more) __builtin_amdgcn_s_wait_tensorcnt(2);  // trivial pass on waves 1..3
    else      __builtin_amdgcn_s_wait_tensorcnt(0);
    __syncthreads();  // publish buf

    v16h af[4];
#pragma unroll
    for (int i = 0; i < 4; ++i)
      af[i] = load_a_frag(&As[buf][mw + 16 * i][0], GSTR, lane);
#pragma unroll
    for (int t = 0; t < 4; ++t) {
      const v16h b = load_b_frag_nk(&Bs[buf][nw + 16 * t][0], GSTR, lane);
#pragma unroll
      for (int i = 0; i < 4; ++i)
        acc[i * 4 + t] = __builtin_amdgcn_wmma_f32_16x16x32_f16(
            false, af[i], false, b, (short)0, acc[i * 4 + t], false, false);
    }
  }

  const int col  = lane & 15;
  const int half = lane >> 4;
#pragma unroll
  for (int i = 0; i < 4; ++i) {
#pragma unroll
    for (int r = 0; r < 8; ++r) {
      const int row = m0 + mw + 16 * i + r + 8 * half;
#pragma unroll
      for (int t = 0; t < 4; ++t) {
        const int cg = n0 + nw + 16 * t + col;
        C[(size_t)row * N + cg] = acc[i * 4 + t][r] + bias[cg];
      }
    }
  }
}

// ---------------------------------------------------------------------------
// Stage 2: per-token RMS-norm + gains + 3D RoPE.
//   qh/kh: head-major f16 [(head*L + l)*HD + d]; q pre-scaled by 1/sqrt(HD).
//   vt:    transposed f16 [(head*HD + d)*L + l]  (TDM-friendly for P@V).
// ---------------------------------------------------------------------------
__global__ __launch_bounds__(256) void norm_rope(
    const float* __restrict__ qf, const float* __restrict__ kf,
    const float* __restrict__ vf, const float* __restrict__ gq,
    const float* __restrict__ gk, const float* __restrict__ cosT,
    const float* __restrict__ sinT, _Float16* __restrict__ qh,
    _Float16* __restrict__ kh, _Float16* __restrict__ vt) {
  const int l   = blockIdx.x;
  const int tid = threadIdx.x;
  const float* qrow = qf + (size_t)l * DIM;
  const float* krow = kf + (size_t)l * DIM;
  const float* vrow = vf + (size_t)l * DIM;
  const float scale = 0.08838834764831845f;  // 1/sqrt(128), folded into Q

  float sq = 0.f, sk = 0.f;
  for (int j = tid; j < DIM; j += 256) {
    const float a = qrow[j]; sq += a * a;
    const float b = krow[j]; sk += b * b;
  }
#pragma unroll
  for (int off = 16; off > 0; off >>= 1) {
    sq += __shfl_xor(sq, off, 32);
    sk += __shfl_xor(sk, off, 32);
  }
  __shared__ float rs[2][8];
  if ((tid & 31) == 0) { rs[0][tid >> 5] = sq; rs[1][tid >> 5] = sk; }
  __syncthreads();
  float tq = 0.f, tk = 0.f;
#pragma unroll
  for (int i = 0; i < 8; ++i) { tq += rs[0][i]; tk += rs[1][i]; }
  const float invq = rsqrtf(tq * (1.0f / DIM) + EPS) * scale;
  const float invk = rsqrtf(tk * (1.0f / DIM) + EPS);

  const int fi  = l / (H_ * W_);
  const int rem = l % (H_ * W_);
  const int hi  = rem / W_;
  const int wi  = rem % W_;

  for (int p = tid; p < DIM / 2; p += 256) {
    const int head = p >> 6;
    const int j    = p & 63;
    const int row  = (j < 22) ? fi : ((j < 43) ? hi : wi);
    const float cv = cosT[row * 64 + j];
    const float sv = sinT[row * 64 + j];
    const int d0 = head * HD + 2 * j;
    const size_t ob = ((size_t)head * LQ + l) * HD + 2 * j;

    const float qr = qrow[d0] * invq * gq[d0];
    const float qi = qrow[d0 + 1] * invq * gq[d0 + 1];
    qh[ob]     = (_Float16)(qr * cv - qi * sv);
    qh[ob + 1] = (_Float16)(qr * sv + qi * cv);

    const float kr = krow[d0] * invk * gk[d0];
    const float ki = krow[d0 + 1] * invk * gk[d0 + 1];
    kh[ob]     = (_Float16)(kr * cv - ki * sv);
    kh[ob + 1] = (_Float16)(kr * sv + ki * cv);
  }
  for (int j = tid; j < DIM; j += 256) {
    const int head = j >> 7;
    const int d    = j & 127;
    vt[((size_t)head * HD + d) * LQ + l] = (_Float16)vrow[j];
  }
}

// ---------------------------------------------------------------------------
// Stage 3: flash attention, TDM double-buffered K / V^T tiles, dynamic LDS.
//   LDS layout (halves):
//     Qs     [64][136]        @ 0        (8704)
//     Ks[2]  [64][136]        @ 8704     (2*8704)
//     Vt[2]  [128][72]        @ 26112    (2*9216)
//     Ps     [4][16][72]      @ 44544    (4608)     total 49152 halves = 96KB
// ---------------------------------------------------------------------------
#define QSTR 136  // 128 data + 8 pad halves (TDM: interval 64DW, amount 4DW)
#define VSTR 72   // 64 data + 8 pad halves  (TDM: interval 32DW, amount 4DW)

__global__ __launch_bounds__(128) void flash_attn(
    const _Float16* __restrict__ Qh, const _Float16* __restrict__ Kh,
    const _Float16* __restrict__ Vt, _Float16* __restrict__ Oh) {
  extern __shared__ _Float16 smem[];
  _Float16* Qs  = smem;                      // 64 x 136
  _Float16* KsB = smem + 8704;               // 2 x 64 x 136
  _Float16* VtB = smem + 26112;              // 2 x 128 x 72
  _Float16* Ps  = smem + 44544;              // 4 x 16 x 72

  const int head = blockIdx.y;
  const int q0   = blockIdx.x * 64;
  const int tid  = threadIdx.x;
  const int lane = tid & 31;
  const int wave = tid >> 5;

  const _Float16* Kbase  = Kh + (size_t)head * LQ * HD;
  const _Float16* Vtbase = Vt + (size_t)head * HD * LQ;

  if (wave == 0) {
    tdm_load_2d_f16(lds_off_of(Qs), Qh + ((size_t)head * LQ + q0) * HD, HD, 64, HD, 5, 3);
    tdm_load_2d_f16(lds_off_of(KsB), Kbase, HD, 64, HD, 5, 3);
    tdm_load_2d_f16(lds_off_of(VtB), Vtbase, 64, HD, LQ, 4, 3);
  }

  float m_st[8], l_st[8];
#pragma unroll
  for (int r = 0; r < 8; ++r) { m_st[r] = -1e30f; l_st[r] = 0.f; }
  v8f acc[8] = {};

  int buf = 0;
  for (int j0 = 0; j0 < LQ; j0 += 64, buf ^= 1) {
    const bool more = (j0 + 64) < LQ;
    __syncthreads();  // buf^1 free to overwrite
    if (wave == 0 && more) {
      tdm_load_2d_f16(lds_off_of(KsB + (buf ^ 1) * 8704),
                      Kbase + (size_t)(j0 + 64) * HD, HD, 64, HD, 5, 3);
      tdm_load_2d_f16(lds_off_of(VtB + (buf ^ 1) * 9216),
                      Vtbase + (j0 + 64), 64, HD, LQ, 4, 3);
    }
    if (more) __builtin_amdgcn_s_wait_tensorcnt(2);
    else      __builtin_amdgcn_s_wait_tensorcnt(0);
    __syncthreads();  // publish buf

    const _Float16* Kt = KsB + buf * 8704;
    const _Float16* Vv = VtB + buf * 9216;

    // ---- S = Q K^T : wave rows [16*wave,+16) x key cols [0,64) ----
    v16h aq[4];
#pragma unroll
    for (int kc = 0; kc < 4; ++kc)
      aq[kc] = load_a_frag(Qs + (16 * wave) * QSTR + 32 * kc, QSTR, lane);

    v8f s[4] = {};
#pragma unroll
    for (int t = 0; t < 4; ++t) {
#pragma unroll
      for (int kc = 0; kc < 4; ++kc) {
        const v16h b = load_b_frag_nk(Kt + (16 * t) * QSTR + 32 * kc, QSTR, lane);
        s[t] = __builtin_amdgcn_wmma_f32_16x16x32_f16(
            false, aq[kc], false, b, (short)0, s[t], false, false);
      }
    }

    // ---- online softmax (row r+8*(lane>=16); cols across 16 lanes x 4 tiles)
#pragma unroll
    for (int r = 0; r < 8; ++r) {
      float mv = -1e30f;
#pragma unroll
      for (int t = 0; t < 4; ++t) mv = fmaxf(mv, s[t][r]);
#pragma unroll
      for (int off = 1; off < 16; off <<= 1) mv = fmaxf(mv, __shfl_xor(mv, off, 32));
      const float mnew = fmaxf(m_st[r], mv);
      float ps = 0.f;
#pragma unroll
      for (int t = 0; t < 4; ++t) {
        const float pe = __expf(s[t][r] - mnew);
        s[t][r] = pe;
        ps += pe;
      }
#pragma unroll
      for (int off = 1; off < 16; off <<= 1) ps += __shfl_xor(ps, off, 32);
      const float alpha = __expf(m_st[r] - mnew);
      l_st[r] = l_st[r] * alpha + ps;
      m_st[r] = mnew;
#pragma unroll
      for (int dt = 0; dt < 8; ++dt) acc[dt][r] *= alpha;
    }

    // ---- stage P (D-layout) into per-wave LDS, re-read as A fragments ----
    _Float16* Pw = Ps + wave * (16 * VSTR);
    {
      const int half = lane >> 4;
      const int col  = lane & 15;
#pragma unroll
      for (int r = 0; r < 8; ++r)
#pragma unroll
        for (int t = 0; t < 4; ++t)
          Pw[(r + 8 * half) * VSTR + 16 * t + col] = (_Float16)s[t][r];
    }

    // ---- O += P V : B fragments from V^T (n-major, coalesced) ----
#pragma unroll
    for (int kc = 0; kc < 2; ++kc) {
      const v16h ap = load_a_frag(Pw + 32 * kc, VSTR, lane);
#pragma unroll
      for (int dt = 0; dt < 8; ++dt) {
        const v16h bv = load_b_frag_nk(Vv + (16 * dt) * VSTR + 32 * kc, VSTR, lane);
        acc[dt] = __builtin_amdgcn_wmma_f32_16x16x32_f16(
            false, ap, false, bv, (short)0, acc[dt], false, false);
      }
    }
  }

  const int half = lane >> 4;
  const int col  = lane & 15;
#pragma unroll
  for (int r = 0; r < 8; ++r) {
    const float inv = 1.0f / l_st[r];
    const int lg = q0 + 16 * wave + r + 8 * half;
#pragma unroll
    for (int dt = 0; dt < 8; ++dt)
      Oh[(size_t)lg * DIM + head * HD + 16 * dt + col] = (_Float16)(acc[dt][r] * inv);
  }
}

// ---------------------------------------------------------------------------
// Launch
// ---------------------------------------------------------------------------
extern "C" void kernel_launch(void* const* d_in, const int* in_sizes, int n_in,
                              void* d_out, int out_size, void* d_ws, size_t ws_size,
                              hipStream_t stream) {
  const float* x    = (const float*)d_in[0];
  const float* fcos = (const float*)d_in[1];
  const float* fsin = (const float*)d_in[2];
  const float* Wq   = (const float*)d_in[3];
  const float* bq   = (const float*)d_in[4];
  const float* Wk   = (const float*)d_in[5];
  const float* bk   = (const float*)d_in[6];
  const float* Wv   = (const float*)d_in[7];
  const float* bv   = (const float*)d_in[8];
  const float* Wo   = (const float*)d_in[9];
  const float* bo   = (const float*)d_in[10];
  const float* gq   = (const float*)d_in[11];
  const float* gk   = (const float*)d_in[12];
  float* out = (float*)d_out;

  const size_t LD = (size_t)LQ * DIM;   // 5,505,024
  const size_t WD = (size_t)DIM * DIM;  // 2,359,296
  char* ws = (char*)d_ws;
  _Float16* x16  = (_Float16*)ws;
  _Float16* wq16 = (_Float16*)(ws + LD * 2);
  _Float16* wk16 = (_Float16*)(ws + LD * 2 + WD * 2);
  _Float16* wv16 = (_Float16*)(ws + LD * 2 + WD * 4);
  _Float16* wo16 = (_Float16*)(ws + LD * 2 + WD * 6);
  float*    qf   = (float*)(ws + LD * 2 + WD * 8);
  float*    kf   = (float*)(ws + LD * 6 + WD * 8);
  float*    vf   = (float*)(ws + LD * 10 + WD * 8);
  _Float16* qh   = (_Float16*)(ws + LD * 14 + WD * 8);
  _Float16* kh   = (_Float16*)(ws + LD * 16 + WD * 8);
  _Float16* vt   = (_Float16*)(ws + LD * 18 + WD * 8);
  _Float16* oh   = x16;  // x16 dead after QKV GEMMs

  // Stage 0: one-time f16 conversions
  cvt_f32_f16<<<2048, 256, 0, stream>>>(x, x16, (int)LD);
  cvt_f32_f16<<<1024, 256, 0, stream>>>(Wq, wq16, (int)WD);
  cvt_f32_f16<<<1024, 256, 0, stream>>>(Wk, wk16, (int)WD);
  cvt_f32_f16<<<1024, 256, 0, stream>>>(Wv, wv16, (int)WD);
  cvt_f32_f16<<<1024, 256, 0, stream>>>(Wo, wo16, (int)WD);

  // Stage 1: Q/K/V projections
  dim3 ggrid(LQ / 128, DIM / 128);
  gemm_f16<<<ggrid, 128, 0, stream>>>(x16, wq16, bq, qf, LQ, DIM, DIM);
  gemm_f16<<<ggrid, 128, 0, stream>>>(x16, wk16, bk, kf, LQ, DIM, DIM);
  gemm_f16<<<ggrid, 128, 0, stream>>>(x16, wv16, bv, vf, LQ, DIM, DIM);

  // Stage 2: RMS-norm + RoPE -> head-major f16 (Q pre-scaled), V transposed
  norm_rope<<<LQ, 256, 0, stream>>>(qf, kf, vf, gq, gk, fcos, fsin, qh, kh, vt);

  // Stage 3: flash attention (96KB dynamic LDS: Q + 2x(K,V^T) + P)
  dim3 fgrid(LQ / 64, HEADS);
  flash_attn<<<fgrid, 128, 98304, stream>>>(qh, kh, vt, oh);

  // Stage 4: output projection -> d_out (f32)
  gemm_f16<<<ggrid, 128, 0, stream>>>(oh, wo16, bo, out, LQ, DIM, DIM);
}
